// FactorGraphGRU_68126771249442
// MI455X (gfx1250) — matrix operand order repaired and checked by reading, not compiled, not run on hardware
//
#include <hip/hip_runtime.h>
#include <hip/hip_bf16.h>
#include <math.h>

#define NN   4096
#define HH   256
#define GG   768   // 3*HH
#define HEADS 2
#define LALPHA 0.2f
#define NEG_BIG -9e15f

typedef __attribute__((ext_vector_type(16))) __bf16 v16bf;
typedef __attribute__((ext_vector_type(8)))  __bf16 v8bf;
typedef __attribute__((ext_vector_type(8)))  float  v8f;

union FragAB { v8bf h[2]; v16bf v; };
union FragC  { float f[8]; v8f  v; };
union PackBF { __bf16 b[2]; unsigned int u; };

enum { MODE_PLAIN = 0, MODE_SIGNED_ND = 1, MODE_POS_ND = 2, MODE_NEG_ND = 3 };

// ---------------------------------------------------------------------------
// WMMA GEMM: C[M x Nd] (f32) = op(A)[M x K] (f32->bf16) @ B[K x Nd] (bf16)
// Block tile 128x128, K-step 32, 256 threads = 8 wave32 (4x2), each wave 2x4
// 16x16 tiles of v_wmma_f32_16x16x32_bf16 (f32 accumulate).
//
// - LDS double-buffered; next tile prefetched into registers during compute.
// - Fragment layouts give 2x ds_load_b128 per fragment:
//     As[row][k]   (stride 32): A frag = runs [half*8,+8) & [16+half*8,+8)
//     Bs_t[col][k] (stride 32): B frag = runs [half*8,+8) & [16+half*8,+8)
// - All 6 fragments loaded before the 8-WMMA clause (single dscnt wait).
// - 32-bit index arithmetic throughout (all tensors < 2^31 elements).
// Requires M%128==0, Nd%128==0, K%32==0 (true for all uses here).
// ---------------------------------------------------------------------------
template <int MODE>
__global__ __launch_bounds__(256)
void gemm_bf16_wmma(const float* __restrict__ A, const __bf16* __restrict__ B,
                    float* __restrict__ C, int M, int K, int Nd) {
  __shared__ __bf16 As[2 * 128 * 32];   // [buf][row][k]
  __shared__ __bf16 Bs[2 * 128 * 32];   // [buf][col][k] (transposed tile)

  const int tid   = threadIdx.x;
  const int lane  = tid & 31;
  const int wave  = tid >> 5;
  const int waveM = wave >> 1;        // 0..3  -> 32 rows each
  const int waveN = wave & 1;         // 0..1  -> 64 cols each
  const int half  = (lane >> 4) & 1;  // lane half
  const int l15   = lane & 15;
  const int rowBase = blockIdx.x * 128;
  const int colBase = blockIdx.y * 128;
  const int steps = K >> 5;

  FragC acc[2][4];
#pragma unroll
  for (int mt = 0; mt < 2; ++mt)
#pragma unroll
    for (int nt = 0; nt < 4; ++nt)
#pragma unroll
      for (int r = 0; r < 8; ++r) acc[mt][nt].f[r] = 0.0f;

  float2       aReg[8];
  unsigned int bReg[8];

  // ---- prologue: fetch tile 0 and stage into buffer 0 ----
#pragma unroll
  for (int j = 0; j < 8; ++j) {
    int i = tid + j * 256;                       // 128x16 float2 elements
    int r = i >> 4, c2 = (i & 15) * 2;
    aReg[j] = *(const float2*)(A + (rowBase + r) * K + c2);
  }
#pragma unroll
  for (int j = 0; j < 8; ++j) {
    int i = tid + j * 256;                       // 32x64 dword elements
    int k = i >> 6, cp = (i & 63) * 2;
    bReg[j] = *(const unsigned int*)(B + k * Nd + colBase + cp);
  }
  {
#pragma unroll
    for (int j = 0; j < 8; ++j) {
      int i = tid + j * 256;
      int r = i >> 4, c2 = (i & 15) * 2;
      int gr = rowBase + r, gk = c2;
      float a0 = aReg[j].x, a1 = aReg[j].y;
      if (MODE == MODE_SIGNED_ND) {
        if (gr == gk)     a0 = 0.0f;
        if (gr == gk + 1) a1 = 0.0f;
      } else if (MODE == MODE_POS_ND) {
        a0 = (a0 > 0.0f && gr != gk)     ? 1.0f : 0.0f;
        a1 = (a1 > 0.0f && gr != gk + 1) ? 1.0f : 0.0f;
      } else if (MODE == MODE_NEG_ND) {
        a0 = (a0 < 0.0f && gr != gk)     ? 1.0f : 0.0f;
        a1 = (a1 < 0.0f && gr != gk + 1) ? 1.0f : 0.0f;
      }
      PackBF p; p.b[0] = (__bf16)a0; p.b[1] = (__bf16)a1;
      *(unsigned int*)(&As[r * 32 + c2]) = p.u;
    }
#pragma unroll
    for (int j = 0; j < 8; ++j) {
      int i = tid + j * 256;
      int k = i >> 6, cp = (i & 63) * 2;
      PackBF p; p.u = bReg[j];
      Bs[(cp + 0) * 32 + k] = p.b[0];
      Bs[(cp + 1) * 32 + k] = p.b[1];
    }
  }
  __syncthreads();

  for (int kt = 0; kt < steps; ++kt) {
    const int cur = kt & 1;
    const int nk0 = (kt + 1) << 5;

    // ---- prefetch next tile into registers (overlaps with WMMA below) ----
    if (kt + 1 < steps) {
#pragma unroll
      for (int j = 0; j < 8; ++j) {
        int i = tid + j * 256;
        int r = i >> 4, c2 = (i & 15) * 2;
        aReg[j] = *(const float2*)(A + (rowBase + r) * K + nk0 + c2);
      }
#pragma unroll
      for (int j = 0; j < 8; ++j) {
        int i = tid + j * 256;
        int k = i >> 6, cp = (i & 63) * 2;
        bReg[j] = *(const unsigned int*)(B + (nk0 + k) * Nd + colBase + cp);
      }
    }

    // ---- compute from buffer `cur`: load all 6 fragments, then 8 WMMAs ----
    {
      const __bf16* Ab = As + cur * (128 * 32);
      const __bf16* Bb = Bs + cur * (128 * 32);
      FragAB af[2], bf_[4];
#pragma unroll
      for (int mt = 0; mt < 2; ++mt) {
        int abase = (waveM * 32 + mt * 16 + l15) * 32 + half * 8;
        af[mt].h[0] = *(const v8bf*)(&Ab[abase]);
        af[mt].h[1] = *(const v8bf*)(&Ab[abase + 16]);
      }
#pragma unroll
      for (int nt = 0; nt < 4; ++nt) {
        int bbase = (waveN * 64 + nt * 16 + l15) * 32 + half * 8;
        bf_[nt].h[0] = *(const v8bf*)(&Bb[bbase]);
        bf_[nt].h[1] = *(const v8bf*)(&Bb[bbase + 16]);
      }
#pragma unroll
      for (int nt = 0; nt < 4; ++nt)
#pragma unroll
        for (int mt = 0; mt < 2; ++mt)
          acc[mt][nt].v = __builtin_amdgcn_wmma_f32_16x16x32_bf16(
              false, af[mt].v, false, bf_[nt].v, (short)0, acc[mt][nt].v,
              false, false);
    }

    // ---- stage prefetched tile into the other buffer ----
    if (kt + 1 < steps) {
      __bf16* An = As + (cur ^ 1) * (128 * 32);
      __bf16* Bn = Bs + (cur ^ 1) * (128 * 32);
#pragma unroll
      for (int j = 0; j < 8; ++j) {
        int i = tid + j * 256;
        int r = i >> 4, c2 = (i & 15) * 2;
        int gr = rowBase + r, gk = nk0 + c2;
        float a0 = aReg[j].x, a1 = aReg[j].y;
        if (MODE == MODE_SIGNED_ND) {
          if (gr == gk)     a0 = 0.0f;
          if (gr == gk + 1) a1 = 0.0f;
        } else if (MODE == MODE_POS_ND) {
          a0 = (a0 > 0.0f && gr != gk)     ? 1.0f : 0.0f;
          a1 = (a1 > 0.0f && gr != gk + 1) ? 1.0f : 0.0f;
        } else if (MODE == MODE_NEG_ND) {
          a0 = (a0 < 0.0f && gr != gk)     ? 1.0f : 0.0f;
          a1 = (a1 < 0.0f && gr != gk + 1) ? 1.0f : 0.0f;
        }
        PackBF p; p.b[0] = (__bf16)a0; p.b[1] = (__bf16)a1;
        *(unsigned int*)(&An[r * 32 + c2]) = p.u;
      }
#pragma unroll
      for (int j = 0; j < 8; ++j) {
        int i = tid + j * 256;
        int k = i >> 6, cp = (i & 63) * 2;
        PackBF p; p.u = bReg[j];
        Bn[(cp + 0) * 32 + k] = p.b[0];
        Bn[(cp + 1) * 32 + k] = p.b[1];
      }
      __syncthreads();
    }
  }

  // ---- C epilogue: VGPR r holds M = r + half*8 (16x16 f32 layout) ----
#pragma unroll
  for (int mt = 0; mt < 2; ++mt) {
#pragma unroll
    for (int nt = 0; nt < 4; ++nt) {
      int gc   = colBase + waveN * 64 + nt * 16 + l15;
      int gr0  = rowBase + waveM * 32 + mt * 16 + half * 8;
      int base = gr0 * Nd + gc;
#pragma unroll
      for (int r = 0; r < 8; ++r)
        C[base + r * Nd] = acc[mt][nt].f[r];
    }
  }
}

// ---------------------------------------------------------------------------
// Helpers: conversions, GAT pieces, GRU, combine
// ---------------------------------------------------------------------------
__global__ void cvt_bf16_kernel(const float* __restrict__ in,
                                __bf16* __restrict__ out, long n) {
  long i = (long)blockIdx.x * blockDim.x + threadIdx.x;
  if (i < n) out[i] = (__bf16)in[i];
}

// in: [R x C] f32 -> out: [C x R] bf16 (for W^T used as GEMM B operand)
__global__ void cvt_bf16_tr_kernel(const float* __restrict__ in,
                                   __bf16* __restrict__ out, int R, int C) {
  long i = (long)blockIdx.x * blockDim.x + threadIdx.x;
  if (i < (long)R * C) {
    int r = (int)(i / C), c = (int)(i % C);
    out[(long)c * R + r] = (__bf16)in[i];
  }
}

// e[i] = leaky_relu( sum_c |hp[i,c]-hm[i,c]| * a[c] ), one block (256=HH) per row
__global__ __launch_bounds__(256)
void gat_e_kernel(const float* __restrict__ hp, const float* __restrict__ hm,
                  const float* __restrict__ a, float* __restrict__ e) {
  __shared__ float red[256];
  const int i = blockIdx.x, c = threadIdx.x;
  red[c] = fabsf(hp[(long)i * HH + c] - hm[(long)i * HH + c]) * a[c];
  __syncthreads();
  for (int s = 128; s > 0; s >>= 1) {
    if (c < s) red[c] += red[c + s];
    __syncthreads();
  }
  if (c == 0) {
    float v = red[0];
    e[i] = v > 0.0f ? v : LALPHA * v;
  }
}

// Row-wise masked softmax producing the full attention matrix.
// scores[i,j] = (edge[i,j]!=0 && i!=j) ? e[j] : NEG_BIG ; softmax over j.
__global__ __launch_bounds__(256)
void attention_rows_kernel(const float* __restrict__ edge,
                           const float* __restrict__ e,
                           float* __restrict__ att) {
  __shared__ float srow[NN];
  __shared__ float red[256];
  const int i = blockIdx.x;
  const int tid = threadIdx.x;

  float m = -3.4e38f;
  for (int j = tid; j < NN; j += 256) {
    float ed = edge[(long)i * NN + j];
    float s = (ed != 0.0f && j != i) ? e[j] : NEG_BIG;
    srow[j] = s;
    m = fmaxf(m, s);
  }
  red[tid] = m; __syncthreads();
  for (int s = 128; s > 0; s >>= 1) {
    if (tid < s) red[tid] = fmaxf(red[tid], red[tid + s]);
    __syncthreads();
  }
  const float mx = red[0]; __syncthreads();

  float sum = 0.0f;
  for (int j = tid; j < NN; j += 256) sum += expf(srow[j] - mx);
  red[tid] = sum; __syncthreads();
  for (int s = 128; s > 0; s >>= 1) {
    if (tid < s) red[tid] += red[tid + s];
    __syncthreads();
  }
  const float inv = 1.0f / red[0]; __syncthreads();

  for (int j = tid; j < NN; j += 256)
    att[(long)i * NN + j] = expf(srow[j] - mx) * inv;
}

__global__ void gru_elem_kernel(const float* __restrict__ gi,
                                const float* __restrict__ gh,
                                const float* __restrict__ bi,
                                const float* __restrict__ bh,
                                const float* __restrict__ h,
                                float* __restrict__ out) {
  long idx = (long)blockIdx.x * blockDim.x + threadIdx.x;
  if (idx >= (long)NN * HH) return;
  int n = (int)(idx >> 8);
  int c = (int)(idx & 255);
  const float* gin = gi + (long)n * GG;
  const float* ghn = gh + (long)n * GG;
  float ir = gin[c]          + bi[c],          hr = ghn[c]          + bh[c];
  float iz = gin[c + HH]     + bi[c + HH],     hz = ghn[c + HH]     + bh[c + HH];
  float in_ = gin[c + 2*HH]  + bi[c + 2*HH],   hn = ghn[c + 2*HH]   + bh[c + 2*HH];
  float r = 1.0f / (1.0f + expf(-(ir + hr)));
  float z = 1.0f / (1.0f + expf(-(iz + hz)));
  float ng = tanhf(in_ + r * hn);
  out[idx] = (1.0f - z) * ng + z * h[idx];
}

__global__ void avg2_kernel(const float* __restrict__ a,
                            const float* __restrict__ b,
                            float* __restrict__ o, long n) {
  long i = (long)blockIdx.x * blockDim.x + threadIdx.x;
  if (i < n) o[i] = 0.5f * (a[i] + b[i]);
}

__global__ void combine_kernel(const float* __restrict__ node_adj,
                               const float* __restrict__ edge_adj,
                               const float* __restrict__ eo,
                               const float* __restrict__ no,
                               float* __restrict__ out) {
  long idx = (long)blockIdx.x * blockDim.x + threadIdx.x;
  if (idx >= (long)NN * HH) return;
  int n = (int)(idx >> 8);
  float dn = node_adj[(long)n * NN + n];
  float de = edge_adj[(long)n * NN + n];
  out[idx] = de * eo[idx] + dn * no[idx];
}

// ---------------------------------------------------------------------------
// Host-side orchestration
// ---------------------------------------------------------------------------
extern "C" void kernel_launch(void* const* d_in, const int* in_sizes, int n_in,
                              void* d_out, int out_size, void* d_ws, size_t ws_size,
                              hipStream_t stream) {
  (void)in_sizes; (void)n_in; (void)out_size; (void)ws_size;

  const float* h        = (const float*)d_in[0];
  const float* node_adj = (const float*)d_in[1];
  const float* edge_adj = (const float*)d_in[2];
  const float* W_att    = (const float*)d_in[3];
  const float* a_vec    = (const float*)d_in[4];
  const float* w_ih_e   = (const float*)d_in[5];
  const float* w_hh_e   = (const float*)d_in[6];
  const float* b_ih_e   = (const float*)d_in[7];
  const float* b_hh_e   = (const float*)d_in[8];
  const float* w_ih_n   = (const float*)d_in[9];
  const float* w_hh_n   = (const float*)d_in[10];
  const float* b_ih_n   = (const float*)d_in[11];
  const float* b_hh_n   = (const float*)d_in[12];
  float* out = (float*)d_out;

  char* ws = (char*)d_ws;
  size_t off = 0;
  auto alloc = [&](size_t bytes) -> void* {
    void* p = (void*)(ws + off);
    off += (bytes + 255) & ~(size_t)255;
    return p;
  };

  float* f_gi   = (float*)alloc((size_t)NN * GG * 4);
  float* f_gh   = (float*)alloc((size_t)NN * GG * 4);
  float* f_ns   = (float*)alloc((size_t)NN * HH * 4);  // node_support
  float* f_eo   = (float*)alloc((size_t)NN * HH * 4);  // edge_output
  float* f_hatt = (float*)alloc((size_t)NN * HH * 4);
  float* f_hp   = (float*)alloc((size_t)NN * HH * 4);
  float* f_hm   = (float*)alloc((size_t)NN * HH * 4);
  float* f_hp0  = (float*)alloc((size_t)NN * HH * 4);
  float* f_hp1  = (float*)alloc((size_t)NN * HH * 4);
  float* f_es   = (float*)alloc((size_t)NN * HH * 4);  // edge_support
  float* f_no   = (float*)alloc((size_t)NN * HH * 4);  // node_output
  float* f_e    = (float*)alloc((size_t)NN * 4);
  __bf16* b_h    = (__bf16*)alloc((size_t)NN * HH * 2);
  __bf16* b_hatt = (__bf16*)alloc((size_t)NN * HH * 2);
  __bf16* b_watt = (__bf16*)alloc((size_t)HEADS * HH * HH * 2);
  __bf16* b_ihe  = (__bf16*)alloc((size_t)HH * GG * 2);
  __bf16* b_hhe  = (__bf16*)alloc((size_t)HH * GG * 2);
  __bf16* b_ihn  = (__bf16*)alloc((size_t)HH * GG * 2);
  __bf16* b_hhn  = (__bf16*)alloc((size_t)HH * GG * 2);

  const long NH = (long)NN * HH;
  const int EBLK = 256;
  auto eg = [&](long n) { return (int)((n + EBLK - 1) / EBLK); };

  // --- conversions (B operands must be bf16) ---
  cvt_bf16_kernel<<<eg(NH), EBLK, 0, stream>>>(h, b_h, NH);
  cvt_bf16_kernel<<<eg((long)HEADS * HH * HH), EBLK, 0, stream>>>(
      W_att, b_watt, (long)HEADS * HH * HH);
  cvt_bf16_tr_kernel<<<eg((long)GG * HH), EBLK, 0, stream>>>(w_ih_e, b_ihe, GG, HH);
  cvt_bf16_tr_kernel<<<eg((long)GG * HH), EBLK, 0, stream>>>(w_hh_e, b_hhe, GG, HH);
  cvt_bf16_tr_kernel<<<eg((long)GG * HH), EBLK, 0, stream>>>(w_ih_n, b_ihn, GG, HH);
  cvt_bf16_tr_kernel<<<eg((long)GG * HH), EBLK, 0, stream>>>(w_hh_n, b_hhn, GG, HH);

  // --- edge (factor) update: node_support = node_adj_norm @ h ; GRU ---
  gemm_bf16_wmma<MODE_SIGNED_ND><<<dim3(NN / 128, HH / 128), 256, 0, stream>>>(
      node_adj, b_h, f_ns, NN, NN, HH);
  gemm_bf16_wmma<MODE_PLAIN><<<dim3(NN / 128, GG / 128), 256, 0, stream>>>(
      f_ns, b_ihe, f_gi, NN, HH, GG);
  gemm_bf16_wmma<MODE_PLAIN><<<dim3(NN / 128, GG / 128), 256, 0, stream>>>(
      h, b_hhe, f_gh, NN, HH, GG);
  gru_elem_kernel<<<eg(NH), EBLK, 0, stream>>>(f_gi, f_gh, b_ih_e, b_hh_e, h, f_eo);

  // --- node update: multi-head GAT ---
  for (int hd = 0; hd < HEADS; ++hd) {
    gemm_bf16_wmma<MODE_PLAIN><<<dim3(NN / 128, HH / 128), 256, 0, stream>>>(
        h, b_watt + (long)hd * HH * HH, f_hatt, NN, HH, HH);
    cvt_bf16_kernel<<<eg(NH), EBLK, 0, stream>>>(f_hatt, b_hatt, NH);
    gemm_bf16_wmma<MODE_POS_ND><<<dim3(NN / 128, HH / 128), 256, 0, stream>>>(
        node_adj, b_hatt, f_hp, NN, NN, HH);
    gemm_bf16_wmma<MODE_NEG_ND><<<dim3(NN / 128, HH / 128), 256, 0, stream>>>(
        node_adj, b_hatt, f_hm, NN, NN, HH);
    gat_e_kernel<<<NN, 256, 0, stream>>>(f_hp, f_hm, a_vec + (long)hd * HH, f_e);

    float* att = out + NH + (long)hd * NN * NN;  // attention is a kernel output
    attention_rows_kernel<<<NN, 256, 0, stream>>>(edge_adj, f_e, att);

    // h_prime = (attention * edge_adj_norm) @ h ; attention is already masked
    gemm_bf16_wmma<MODE_PLAIN><<<dim3(NN / 128, HH / 128), 256, 0, stream>>>(
        att, b_h, hd == 0 ? f_hp0 : f_hp1, NN, NN, HH);
  }
  avg2_kernel<<<eg(NH), EBLK, 0, stream>>>(f_hp0, f_hp1, f_es, NH);

  gemm_bf16_wmma<MODE_PLAIN><<<dim3(NN / 128, GG / 128), 256, 0, stream>>>(
      f_es, b_ihn, f_gi, NN, HH, GG);
  gemm_bf16_wmma<MODE_PLAIN><<<dim3(NN / 128, GG / 128), 256, 0, stream>>>(
      h, b_hhn, f_gh, NN, HH, GG);
  gru_elem_kernel<<<eg(NH), EBLK, 0, stream>>>(f_gi, f_gh, b_ih_n, b_hh_n, h, f_no);

  // --- diagonal gating combine ---
  combine_kernel<<<eg(NH), EBLK, 0, stream>>>(node_adj, edge_adj, f_eo, f_no, out);
}
